// VSSBlock_85401129714116
// MI455X (gfx1250) — compile-verified, matrix operand order checked by reading, not run on hardware
//
#include <hip/hip_runtime.h>

typedef __attribute__((ext_vector_type(16))) __bf16 v16bf;
typedef __attribute__((ext_vector_type(8)))  __bf16 v8bf;
typedef __attribute__((ext_vector_type(8)))  float  v8f;

#define BATCH 32
#define HH 56
#define WW 56
#define CM 96          // d_model
#define DI 192         // d_inner
#define TILE 8         // 8x8 spatial tile per workgroup
#define HALO 10        // (TILE + 2)
#define NPOS 100       // HALO*HALO
#define MPAD 112       // padded to 7 * 16 rows
#define NB1_TILES 36   // GEMM1: 3 ktiles * 12 ntiles
#define NB2_TILES_PER_TAP 36 // conv: 6 ktiles * 6 ntiles
#define B1_ELEMS (NB1_TILES * 512)             // 18432 bf16
#define B2_ELEMS (9 * NB2_TILES_PER_TAP * 512) // 165888 bf16

// ---------------------------------------------------------------------------
// Prepass: pack w_in (first 192 cols) and conv_w (9 taps of 192x96) as bf16
// B-fragments. Fragment layout (per 16x(K=32) tile, per lane, i = 0..15):
//   n = nt*16 + (lane & 15)
//   k = kt*32 + (lane>>4)*8 + (i & 7) + (i>>3)*16
// so each lane's 16 values are one contiguous 32-byte chunk in memory.
// ---------------------------------------------------------------------------
__global__ __launch_bounds__(256) void pack_weights(
    const float* __restrict__ w_in, const float* __restrict__ conv_w,
    __bf16* __restrict__ wsB)
{
    int e = blockIdx.x * 256 + threadIdx.x;
    if (e >= B1_ELEMS + B2_ELEMS) return;
    float val;
    if (e < B1_ELEMS) {
        int tile = e >> 9, sub = e & 511;
        int lane = sub >> 4, i = sub & 15;
        int kt = tile / 12, nt = tile % 12;
        int n = nt * 16 + (lane & 15);
        int k = kt * 32 + ((lane >> 4) * 8) + (i & 7) + ((i >> 3) * 16);
        val = w_in[k * (2 * DI) + n];              // (96, 384), cols 0..191
    } else {
        int e2   = e - B1_ELEMS;
        int tap  = e2 / B1_ELEMS;                  // 18432 elems per tap
        int rem  = e2 % B1_ELEMS;
        int tile = rem >> 9, sub = rem & 511;
        int lane = sub >> 4, i = sub & 15;
        int kt = tile / 6, nt = tile % 6;
        int n = nt * 16 + (lane & 15);
        int k = kt * 32 + ((lane >> 4) * 8) + (i & 7) + ((i >> 3) * 16);
        val = conv_w[(tap * DI + k) * CM + n];     // (3,3,192,96) flattened
    }
    wsB[e] = (__bf16)val;
}

// ---------------------------------------------------------------------------
// Fused LN -> in_proj GEMM -> 3x3 conv (as 9 shifted GEMMs) -> silu + residual
// ---------------------------------------------------------------------------
__global__ __launch_bounds__(256) void vss_fused(
    const float* __restrict__ x,
    const float* __restrict__ ln_w, const float* __restrict__ ln_b,
    const float* __restrict__ b_in, const float* __restrict__ conv_b,
    const __bf16* __restrict__ wB1, const __bf16* __restrict__ wB2,
    float* __restrict__ out)
{
    __shared__ __bf16 sH[MPAD * CM];    // layernormed input (halo tile), bf16
    __shared__ __bf16 sX1[MPAD * DI];   // x1 = h @ w_in + b_in, bf16

    const int tid  = threadIdx.x;
    const int lane = tid & 31;
    const int wave = tid >> 5;

    const int blk  = blockIdx.x;
    const int tx   = blk % (WW / TILE);
    const int ty   = (blk / (WW / TILE)) % (HH / TILE);
    const int bimg = blk / ((WW / TILE) * (HH / TILE));
    const int gh0  = ty * TILE;
    const int gw0  = tx * TILE;

    // ---------------- Phase 1: LayerNorm into sH (bf16) ----------------
    if (tid < MPAD) {
        const int ph = tid;
        bool valid = false;
        long base = 0;
        if (ph < NPOS) {
            int hr = ph / HALO, hc = ph % HALO;
            int gh = gh0 + hr - 1, gw = gw0 + hc - 1;
            if (gh >= 0 && gh < HH && gw >= 0 && gw < WW) {
                valid = true;
                base = (((long)bimg * HH + gh) * WW + gw) * CM;
            }
        }
        if (valid) {
            float sum = 0.f, sq = 0.f;
            #pragma unroll 4
            for (int c = 0; c < CM; c += 4) {
                float4 v = *(const float4*)&x[base + c];
                sum += v.x + v.y + v.z + v.w;
                sq  += v.x * v.x + v.y * v.y + v.z * v.z + v.w * v.w;
            }
            float mean = sum * (1.f / CM);
            float var  = sq * (1.f / CM) - mean * mean;
            float rs   = rsqrtf(var + 1e-5f);
            #pragma unroll 4
            for (int c = 0; c < CM; ++c) {
                float h = (x[base + c] - mean) * rs * ln_w[c] + ln_b[c];
                sH[ph * CM + c] = (__bf16)h;
            }
        } else {
            v8bf z = {};
            #pragma unroll
            for (int c = 0; c < CM; c += 8)
                *(v8bf*)&sH[ph * CM + c] = z;
        }
    }
    __syncthreads();

    // ------- Phase 2: x1 = h @ w_in + b_in (bf16 WMMA, 4-wide N-blocking) -------
    // M = 112 (7 tiles), N = 192 (3 groups of 4 tiles), K = 96 (3 steps of 32)
    {
        const int mr    = lane & 15;
        const int khalf = (lane >> 4) * 8;
        for (int t = wave; t < 7 * 3; t += 8) {
            int mt = t / 3, ng = (t % 3) * 4;       // n-tile group base
            int mrow = mt * 16 + mr;
            v8f acc[4] = {};
            #pragma unroll
            for (int kt = 0; kt < 3; ++kt) {
                int k0 = kt * 32 + khalf;
                v8bf lo = *(const v8bf*)&sH[mrow * CM + k0];
                v8bf hi = *(const v8bf*)&sH[mrow * CM + k0 + 16];
                v16bf a = __builtin_shufflevector(lo, hi,
                    0,1,2,3,4,5,6,7,8,9,10,11,12,13,14,15);
                #pragma unroll
                for (int q = 0; q < 4; ++q) {
                    v16bf b = *(const v16bf*)
                        &wB1[((kt * 12 + (ng + q)) * 32 + lane) * 16];
                    acc[q] = __builtin_amdgcn_wmma_f32_16x16x32_bf16(
                        false, a, false, b, (short)0, acc[q], false, false);
                }
            }
            int mbase = mt * 16 + khalf;
            #pragma unroll
            for (int q = 0; q < 4; ++q) {
                int n = (ng + q) * 16 + mr;
                float bin = b_in[n];
                #pragma unroll
                for (int j = 0; j < 8; ++j) {
                    float v = acc[q][j] + bin;
                    sX1[(mbase + j) * DI + n] = (__bf16)v;
                }
            }
        }
    }
    __syncthreads();

    // Phase 2b: SAME padding zeroes x1 (not b_in) at out-of-image halo rows
    if (tid < MPAD) {
        const int ph = tid;
        bool oob = true;
        if (ph < NPOS) {
            int hr = ph / HALO, hc = ph % HALO;
            int gh = gh0 + hr - 1, gw = gw0 + hc - 1;
            oob = !(gh >= 0 && gh < HH && gw >= 0 && gw < WW);
        }
        if (oob) {
            v8bf z = {};
            #pragma unroll
            for (int c = 0; c < DI; c += 8)
                *(v8bf*)&sX1[ph * DI + c] = z;
        }
    }
    __syncthreads();

    // ------- Phase 3: y = conv3x3(x1) as 9 shifted GEMMs (3-wide N-blocking) -------
    // M = 64 out positions (4 tiles), N = 96 (2 groups of 3 tiles), K = 192/tap.
    // wave -> (mt = wave&3, ng = (wave>>2)*3): all 8 waves exactly cover 4x6 tiles.
    {
        const int mr    = lane & 15;
        const int khalf = (lane >> 4) * 8;
        const int mt    = wave & 3;
        const int ng    = (wave >> 2) * 3;
        const int ml    = mt * 16 + mr;          // output position 0..63
        const int r     = ml >> 3, cc = ml & 7;
        v8f acc[3] = {};
        for (int tap = 0; tap < 9; ++tap) {
            int dh = tap / 3, dw = tap % 3;
            const __bf16* row = &sX1[((r + dh) * HALO + (cc + dw)) * DI];
            #pragma unroll
            for (int kt = 0; kt < 6; ++kt) {
                int k0 = kt * 32 + khalf;
                v8bf lo = *(const v8bf*)&row[k0];
                v8bf hi = *(const v8bf*)&row[k0 + 16];
                v16bf a = __builtin_shufflevector(lo, hi,
                    0,1,2,3,4,5,6,7,8,9,10,11,12,13,14,15);
                #pragma unroll
                for (int q = 0; q < 3; ++q) {
                    v16bf b = *(const v16bf*)
                        &wB2[(((tap * 6 + kt) * 6 + (ng + q)) * 32 + lane) * 16];
                    acc[q] = __builtin_amdgcn_wmma_f32_16x16x32_bf16(
                        false, a, false, b, (short)0, acc[q], false, false);
                }
            }
        }
        const int mbase = mt * 16 + khalf;
        #pragma unroll
        for (int q = 0; q < 3; ++q) {
            int n = (ng + q) * 16 + mr;
            float cb = conv_b[n];
            #pragma unroll
            for (int j = 0; j < 8; ++j) {
                int p  = mbase + j;
                int rr = p >> 3, pc = p & 7;
                long gi = (((long)bimg * HH + (gh0 + rr)) * WW + (gw0 + pc)) * CM + n;
                float y = acc[q][j] + cb;
                float s = y * (1.f / (1.f + __expf(-y)));   // silu
                out[gi] = x[gi] + s;
            }
        }
    }
}

extern "C" void kernel_launch(void* const* d_in, const int* in_sizes, int n_in,
                              void* d_out, int out_size, void* d_ws, size_t ws_size,
                              hipStream_t stream) {
    (void)in_sizes; (void)n_in; (void)out_size; (void)ws_size;
    const float* x      = (const float*)d_in[0];
    const float* ln_w   = (const float*)d_in[1];
    const float* ln_b   = (const float*)d_in[2];
    const float* w_in   = (const float*)d_in[3];
    const float* b_in   = (const float*)d_in[4];
    const float* conv_w = (const float*)d_in[5];
    const float* conv_b = (const float*)d_in[6];
    float* out = (float*)d_out;
    __bf16* wsB = (__bf16*)d_ws;

    const int total = B1_ELEMS + B2_ELEMS;       // 184320
    pack_weights<<<(total + 255) / 256, 256, 0, stream>>>(w_in, conv_w, wsB);

    const int nblocks = BATCH * (HH / TILE) * (WW / TILE); // 1568
    vss_fused<<<nblocks, 256, 0, stream>>>(x, ln_w, ln_b, b_in, conv_b,
                                           wsB, wsB + B1_ELEMS, out);
}
